// MLAP_GIN_58042188038188
// MI455X (gfx1250) — matrix operand order repaired and checked by reading, not compiled, not run on hardware
//
#include <hip/hip_runtime.h>
#include <math.h>

#define NN   100000
#define EE   1600000
#define GG   1000
#define DIMH 256
#define NDEP 4

typedef __attribute__((ext_vector_type(16))) __bf16 v16bf;
typedef __attribute__((ext_vector_type(8)))  float  v8f;

union BFrag { unsigned u[8]; v16bf v; };

// ---------------------------------------------------------------- utilities
__global__ void fill_f32(float* __restrict__ p, float v, size_t n) {
    size_t i = (size_t)blockIdx.x * blockDim.x + threadIdx.x;
    if (i < n) p[i] = v;
}

__global__ void copy_f32x4(const float* __restrict__ s, float* __restrict__ d, size_t n4) {
    size_t i = (size_t)blockIdx.x * blockDim.x + threadIdx.x;
    if (i < n4) ((float4*)d)[i] = ((const float4*)s)[i];
}

__global__ void cvt_bf16_kernel(const float* __restrict__ s, __bf16* __restrict__ d, size_t n) {
    size_t i = (size_t)blockIdx.x * blockDim.x + threadIdx.x;
    if (i < n) d[i] = (__bf16)s[i];
}

__device__ __forceinline__ void atomicMaxF(float* addr, float val) {
    int* ai = (int*)addr;
    int old = __float_as_int(*addr);
    while (__int_as_float(old) < val) {
        int prev = atomicCAS(ai, old, __float_as_int(val));
        if (prev == old) break;
        old = prev;
    }
}

// ------------------------------------------------------------ node encoder
__global__ void encode_kernel(const int* __restrict__ xi, const int* __restrict__ nd,
                              const float* __restrict__ en, const float* __restrict__ ed,
                              float* __restrict__ x, float* __restrict__ xin) {
    size_t t = (size_t)blockIdx.x * blockDim.x + threadIdx.x;
    if (t >= (size_t)NN * (DIMH / 4)) return;
    size_t i = t >> 6;
    int    c = (int)(t & 63) * 4;
    float4 a = *(const float4*)(en + (size_t)xi[i] * DIMH + c);
    float4 b = *(const float4*)(ed + (size_t)nd[i] * DIMH + c);
    float4 r; r.x = a.x + b.x; r.y = a.y + b.y; r.z = a.z + b.z; r.w = a.w + b.w;
    *(float4*)(x   + i * DIMH + c) = r;
    *(float4*)(xin + i * DIMH + c) = r;
}

__global__ void count_kernel(const int* __restrict__ batch, float* __restrict__ cnt) {
    int i = blockIdx.x * blockDim.x + threadIdx.x;
    if (i < NN) atomicAdd(cnt + batch[i], 1.0f);
}

// ---------------------------------------------------------- edge aggregate
__global__ void scatter_kernel(const float* __restrict__ x, const int* __restrict__ src,
                               const int* __restrict__ dst, float* __restrict__ h) {
    size_t t = (size_t)blockIdx.x * blockDim.x + threadIdx.x;
    if (t >= (size_t)EE * (DIMH / 4)) return;
    size_t e = t >> 6;
    int    c = (int)(t & 63) * 4;
    int s = src[e], d = dst[e];
    float4 v = *(const float4*)(x + (size_t)s * DIMH + c);
    float* hp = h + (size_t)d * DIMH + c;
    atomicAdd(hp + 0, v.x); atomicAdd(hp + 1, v.y);
    atomicAdd(hp + 2, v.z); atomicAdd(hp + 3, v.w);
}

// --------------------------------------------------------- bf16 WMMA GEMM
// out[rows,ncols] = optionalReLU(A_f32[rows,K] @ W_bf16[K,ncols] + bias)
// block = 8 waves -> 32 rows x 256 cols; wave tile = 16 x 64 (4 wmma tiles)
// A converted fp32->bf16 in-register; W pre-converted to bf16.
__global__ __launch_bounds__(256) void gemm_bf16_wmma(
        const float* __restrict__ A, const __bf16* __restrict__ W,
        const float* __restrict__ bias, float* __restrict__ out,
        int rows, int K, int ncols, int doRelu)
{
    const int lane = threadIdx.x & 31;
    const int wv   = threadIdx.x >> 5;        // 0..7
    const int hl   = lane >> 4;               // half of wave
    const int r    = lane & 15;
    const int rowBase = blockIdx.x * 32 + (wv >> 2) * 16;
    const int colBase = blockIdx.y * 256 + (wv & 3) * 64;

    v8f acc0, acc1, acc2, acc3;
#pragma unroll
    for (int i = 0; i < 8; ++i) { acc0[i] = 0.f; acc1[i] = 0.f; acc2[i] = 0.f; acc3[i] = 0.f; }

    const float* Arow = A + (size_t)(rowBase + r) * K;

    for (int k0 = 0; k0 < K; k0 += 32) {
        // A fragment: 16x32 bf16, ISA layout (lane half splits K, 2 values/VGPR)
        v16bf af;
#pragma unroll
        for (int j = 0; j < 8; ++j) {
            int k = k0 + ((j < 4) ? (hl * 8 + j * 2) : (16 + hl * 8 + (j - 4) * 2));
            float2 f = *(const float2*)(Arow + k);
            af[2 * j + 0] = (__bf16)f.x;
            af[2 * j + 1] = (__bf16)f.y;
        }
        // Four B fragments (32x16 bf16): lane%16 = n, elements = k (same order as A).
        // Loaded as scalar bf16, strided by ncols per k.
        const __bf16* Wrow = W + (size_t)k0 * ncols + colBase + r;
        BFrag b0, b1, b2, b3;
#pragma unroll
        for (int j = 0; j < 8; ++j) {
            int kk = (j < 4) ? (hl * 8 + j * 2) : (16 + hl * 8 + (j - 4) * 2);
            const __bf16* p0 = Wrow + (size_t)kk * ncols;
            const __bf16* p1 = p0 + ncols;
            b0.v[2 * j] = p0[0];  b0.v[2 * j + 1] = p1[0];
            b1.v[2 * j] = p0[16]; b1.v[2 * j + 1] = p1[16];
            b2.v[2 * j] = p0[32]; b2.v[2 * j + 1] = p1[32];
            b3.v[2 * j] = p0[48]; b3.v[2 * j + 1] = p1[48];
        }
        // back-to-back WMMAs: no A/B WAR interleave -> no hazard NOPs needed
        acc0 = __builtin_amdgcn_wmma_f32_16x16x32_bf16(false, af, false, b0.v, (short)0, acc0, false, false);
        acc1 = __builtin_amdgcn_wmma_f32_16x16x32_bf16(false, af, false, b1.v, (short)0, acc1, false, false);
        acc2 = __builtin_amdgcn_wmma_f32_16x16x32_bf16(false, af, false, b2.v, (short)0, acc2, false, false);
        acc3 = __builtin_amdgcn_wmma_f32_16x16x32_bf16(false, af, false, b3.v, (short)0, acc3, false, false);
    }

    // epilogue: bias (+ReLU); D layout: lane half -> M+8, vgpr idx -> M low
    v8f accs[4] = {acc0, acc1, acc2, acc3};
#pragma unroll
    for (int t = 0; t < 4; ++t) {
        int   n = colBase + t * 16 + r;
        float b = bias[n];
#pragma unroll
        for (int i = 0; i < 8; ++i) {
            int m = rowBase + i + hl * 8;
            float v = accs[t][i] + b;
            if (doRelu) v = fmaxf(v, 0.0f);
            if (m < rows) out[(size_t)m * ncols + n] = v;
        }
    }
}

// --------------------------------------------------------------- GraphNorm
__global__ void gn_sum_kernel(const float* __restrict__ x, const int* __restrict__ batch,
                              float* __restrict__ gsum) {
    size_t t = (size_t)blockIdx.x * blockDim.x + threadIdx.x;
    if (t >= (size_t)NN * (DIMH / 4)) return;
    size_t i = t >> 6;
    int    c = (int)(t & 63) * 4;
    int    g = batch[i];
    float4 v = *(const float4*)(x + i * DIMH + c);
    float* p = gsum + (size_t)g * DIMH + c;
    atomicAdd(p + 0, v.x); atomicAdd(p + 1, v.y);
    atomicAdd(p + 2, v.z); atomicAdd(p + 3, v.w);
}

__global__ void gn_center_kernel(const float* __restrict__ x, const int* __restrict__ batch,
                                 const float* __restrict__ gsum, const float* __restrict__ cnt,
                                 const float* __restrict__ alpha,
                                 float* __restrict__ cen, float* __restrict__ gvar) {
    size_t t = (size_t)blockIdx.x * blockDim.x + threadIdx.x;
    if (t >= (size_t)NN * (DIMH / 4)) return;
    size_t i  = t >> 6;
    int    c0 = (int)(t & 63) * 4;
    int    g  = batch[i];
    float  ic = 1.0f / fmaxf(cnt[g], 1.0f);
    float4 v  = *(const float4*)(x + i * DIMH + c0);
    float vv[4] = {v.x, v.y, v.z, v.w};
    float* cp = cen + i * DIMH + c0;
#pragma unroll
    for (int k = 0; k < 4; ++k) {
        int   c  = c0 + k;
        float m  = gsum[(size_t)g * DIMH + c] * ic;
        float ce = vv[k] - alpha[c] * m;
        cp[k] = ce;
        atomicAdd(gvar + (size_t)g * DIMH + c, ce * ce);
    }
}

__global__ void gn_final_kernel(const float* __restrict__ cen, const int* __restrict__ batch,
                                const float* __restrict__ gvar, const float* __restrict__ cnt,
                                const float* __restrict__ gamma, const float* __restrict__ beta,
                                float* __restrict__ x, float* __restrict__ xin, int doRelu) {
    size_t t = (size_t)blockIdx.x * blockDim.x + threadIdx.x;
    if (t >= (size_t)NN * (DIMH / 4)) return;
    size_t i  = t >> 6;
    int    c0 = (int)(t & 63) * 4;
    int    g  = batch[i];
    float  ic = 1.0f / fmaxf(cnt[g], 1.0f);
#pragma unroll
    for (int k = 0; k < 4; ++k) {
        int   c   = c0 + k;
        float var = gvar[(size_t)g * DIMH + c] * ic;
        float inv = rsqrtf(var + 1e-5f);
        float v   = gamma[c] * cen[i * DIMH + c] * inv + beta[c];
        if (doRelu) v = fmaxf(v, 0.0f);
        v += xin[i * DIMH + c];       // residual
        x[i * DIMH + c]   = v;
        xin[i * DIMH + c] = v;        // x_in for next depth
    }
}

// --------------------------------------------------------- attention pool
__global__ void gate_dot_kernel(const float* __restrict__ gh, const float* __restrict__ w2,
                                const float* __restrict__ b2, float* __restrict__ gate) {
    int wid  = blockIdx.x * (blockDim.x >> 5) + (threadIdx.x >> 5);
    int lane = threadIdx.x & 31;
    if (wid >= NN) return;
    const float* row = gh + (size_t)wid * 512;
    float s = 0.0f;
#pragma unroll
    for (int c = lane; c < 512; c += 32) s += row[c] * w2[c];
#pragma unroll
    for (int off = 16; off > 0; off >>= 1) s += __shfl_xor(s, off, 32);
    if (lane == 0) gate[wid] = s + b2[0];
}

__global__ void seg_max_kernel(const float* __restrict__ gate, const int* __restrict__ batch,
                               float* __restrict__ gmax) {
    int i = blockIdx.x * blockDim.x + threadIdx.x;
    if (i < NN) atomicMaxF(gmax + batch[i], gate[i]);
}

__global__ void seg_exp_kernel(const float* __restrict__ gate, const int* __restrict__ batch,
                               const float* __restrict__ gmax, float* __restrict__ ebuf,
                               float* __restrict__ gden) {
    int i = blockIdx.x * blockDim.x + threadIdx.x;
    if (i < NN) {
        int   g = batch[i];
        float e = __expf(gate[i] - gmax[g]);
        ebuf[i] = e;
        atomicAdd(gden + g, e);
    }
}

__global__ void pool_kernel(const float* __restrict__ x, const int* __restrict__ batch,
                            const float* __restrict__ ebuf, const float* __restrict__ gden,
                            float* __restrict__ outd) {
    size_t t = (size_t)blockIdx.x * blockDim.x + threadIdx.x;
    if (t >= (size_t)NN * (DIMH / 4)) return;
    size_t i = t >> 6;
    int    c = (int)(t & 63) * 4;
    int    g = batch[i];
    float  w = ebuf[i] / gden[g];
    float4 v = *(const float4*)(x + i * DIMH + c);
    float* p = outd + (size_t)g * DIMH + c;
    atomicAdd(p + 0, w * v.x); atomicAdd(p + 1, w * v.y);
    atomicAdd(p + 2, w * v.z); atomicAdd(p + 3, w * v.w);
}

__global__ void final_mean_kernel(float* __restrict__ out) {
    int j = blockIdx.x * blockDim.x + threadIdx.x;
    if (j < GG * DIMH) {
        float s = out[j] + out[GG * DIMH + j] + out[2 * GG * DIMH + j] + out[3 * GG * DIMH + j];
        out[4 * GG * DIMH + j] = 0.25f * s;
    }
}

// ------------------------------------------------------------ orchestration
extern "C" void kernel_launch(void* const* d_in, const int* in_sizes, int n_in,
                              void* d_out, int out_size, void* d_ws, size_t ws_size,
                              hipStream_t stream) {
    const int*   x_idx    = (const int*)d_in[0];
    const int*   node_dp  = (const int*)d_in[1];
    const int*   ei       = (const int*)d_in[2];
    const int*   batch    = (const int*)d_in[3];
    const float* emb_node = (const float*)d_in[4];
    const float* emb_dep  = (const float*)d_in[5];
    const float* gin_w1   = (const float*)d_in[6];
    const float* gin_b1   = (const float*)d_in[7];
    const float* gin_w2   = (const float*)d_in[8];
    const float* gin_b2   = (const float*)d_in[9];
    const float* gn_alpha = (const float*)d_in[10];
    const float* gn_gamma = (const float*)d_in[11];
    const float* gn_beta  = (const float*)d_in[12];
    const float* att_w1   = (const float*)d_in[13];
    const float* att_b1   = (const float*)d_in[14];
    const float* att_w2   = (const float*)d_in[15];
    const float* att_b2   = (const float*)d_in[16];
    const int* src = ei;
    const int* dst = ei + EE;
    float* out = (float*)d_out;

    // workspace carving (bh and bmid MUST stay contiguous: [N,512] gate buffer)
    char*  ws  = (char*)d_ws;
    size_t off = 0;
    auto carve = [&](size_t bytes) -> char* {
        char* p = ws + off;
        off += (bytes + 255) & ~(size_t)255;
        return p;
    };
    float*  gsum  = (float*)carve((size_t)GG * DIMH * 4);
    float*  gvar  = (float*)carve((size_t)GG * DIMH * 4);
    float*  cnt   = (float*)carve((size_t)GG * 4);
    float*  gmax  = (float*)carve((size_t)GG * 4);
    float*  gden  = (float*)carve((size_t)GG * 4);
    float*  gate  = (float*)carve((size_t)NN * 4);
    float*  ebuf  = (float*)carve((size_t)NN * 4);
    __bf16* wb_g1 = (__bf16*)carve((size_t)NDEP * DIMH * DIMH * 2);  // gin_w1 bf16
    __bf16* wb_g2 = (__bf16*)carve((size_t)NDEP * DIMH * DIMH * 2);  // gin_w2 bf16
    __bf16* wb_a1 = (__bf16*)carve((size_t)NDEP * DIMH * 512 * 2);   // att_w1 bf16
    float*  bx    = (float*)carve((size_t)NN * DIMH * 4);   // x (fp32 master)
    float*  bxin  = (float*)carve((size_t)NN * DIMH * 4);   // residual input
    float*  bh    = (float*)carve((size_t)NN * DIMH * 4);   // h / x2 / gate-hidden lo
    float*  bmid  = (float*)carve((size_t)NN * DIMH * 4);   // mid / centered / gate-hidden hi

    const int TB = 256;
    auto nb = [&](size_t n) { return (unsigned)((n + TB - 1) / TB); };
    const size_t NC4 = (size_t)NN * (DIMH / 4);
    const size_t EC4 = (size_t)EE * (DIMH / 4);

    // one-time per launch: weight bf16 conversion, output zero, counts, encoder
    cvt_bf16_kernel<<<nb((size_t)NDEP * DIMH * DIMH), TB, 0, stream>>>(gin_w1, wb_g1, (size_t)NDEP * DIMH * DIMH);
    cvt_bf16_kernel<<<nb((size_t)NDEP * DIMH * DIMH), TB, 0, stream>>>(gin_w2, wb_g2, (size_t)NDEP * DIMH * DIMH);
    cvt_bf16_kernel<<<nb((size_t)NDEP * DIMH * 512), TB, 0, stream>>>(att_w1, wb_a1, (size_t)NDEP * DIMH * 512);
    fill_f32<<<nb((size_t)5 * GG * DIMH), TB, 0, stream>>>(out, 0.0f, (size_t)5 * GG * DIMH);
    fill_f32<<<nb(GG), TB, 0, stream>>>(cnt, 0.0f, GG);
    encode_kernel<<<nb(NC4), TB, 0, stream>>>(x_idx, node_dp, emb_node, emb_dep, bx, bxin);
    count_kernel<<<nb(NN), TB, 0, stream>>>(batch, cnt);

    for (int d = 0; d < NDEP; ++d) {
        // h = x + sum_{j->i} x_j
        copy_f32x4<<<nb(NC4), TB, 0, stream>>>(bx, bh, NC4);
        scatter_kernel<<<nb(EC4), TB, 0, stream>>>(bx, src, dst, bh);

        // GIN MLP via WMMA
        gemm_bf16_wmma<<<dim3(NN / 32, 1), 256, 0, stream>>>(
            bh, wb_g1 + (size_t)d * DIMH * DIMH, gin_b1 + d * DIMH, bmid, NN, DIMH, DIMH, 1);
        gemm_bf16_wmma<<<dim3(NN / 32, 1), 256, 0, stream>>>(
            bmid, wb_g2 + (size_t)d * DIMH * DIMH, gin_b2 + d * DIMH, bh, NN, DIMH, DIMH, 0);

        // GraphNorm
        fill_f32<<<nb((size_t)GG * DIMH), TB, 0, stream>>>(gsum, 0.0f, (size_t)GG * DIMH);
        fill_f32<<<nb((size_t)GG * DIMH), TB, 0, stream>>>(gvar, 0.0f, (size_t)GG * DIMH);
        gn_sum_kernel<<<nb(NC4), TB, 0, stream>>>(bh, batch, gsum);
        gn_center_kernel<<<nb(NC4), TB, 0, stream>>>(bh, batch, gsum, cnt,
                                                     gn_alpha + d * DIMH, bmid, gvar);
        gn_final_kernel<<<nb(NC4), TB, 0, stream>>>(bmid, batch, gvar, cnt,
                                                    gn_gamma + d * DIMH, gn_beta + d * DIMH,
                                                    bx, bxin, (d < NDEP - 1) ? 1 : 0);

        // attention-gated pooling: hidden = relu(x @ W1 + b1) -> [N,512] over bh..bmid
        gemm_bf16_wmma<<<dim3(NN / 32, 2), 256, 0, stream>>>(
            bx, wb_a1 + (size_t)d * DIMH * 512, att_b1 + d * 512, bh, NN, DIMH, 512, 1);
        gate_dot_kernel<<<NN / 8, 256, 0, stream>>>(bh, att_w2 + (size_t)d * 512,
                                                    att_b2 + d, gate);
        fill_f32<<<nb(GG), TB, 0, stream>>>(gmax, -INFINITY, GG);
        fill_f32<<<nb(GG), TB, 0, stream>>>(gden, 0.0f, GG);
        seg_max_kernel<<<nb(NN), TB, 0, stream>>>(gate, batch, gmax);
        seg_exp_kernel<<<nb(NN), TB, 0, stream>>>(gate, batch, gmax, ebuf, gden);
        pool_kernel<<<nb(NC4), TB, 0, stream>>>(bx, batch, ebuf, gden,
                                                out + (size_t)d * GG * DIMH);
    }

    final_mean_kernel<<<nb((size_t)GG * DIMH), TB, 0, stream>>>(out);
}
